// minGRU_49830210568836
// MI455X (gfx1250) — compile-verified, matrix operand order checked by reading, not run on hardware
//
#include <hip/hip_runtime.h>

typedef __bf16 bf16;
typedef __attribute__((ext_vector_type(16))) __bf16 v16bf;
typedef __attribute__((ext_vector_type(8)))  __bf16 v8bf;
typedef __attribute__((ext_vector_type(8)))  float  v8f;
typedef __attribute__((ext_vector_type(4)))  float  v4f;
typedef __attribute__((ext_vector_type(4)))  int    v4i;

#define D_IN   1024      // D == E
#define N_OUT  2048      // 2*E
#define BT     32768     // B*T rows
#define TSEQ   4096
#define BATCH  8
#define BM 128
#define BN 128
#define BK 32

// ---------------------------------------------------------------------------
// gfx1250 async global->LDS staging (ASYNCcnt-tracked), with sync fallback.
// Builtin parameter type (from hipcc diagnostic): v4i* in AS1 / AS3.
// ---------------------------------------------------------------------------
#if __has_builtin(__builtin_amdgcn_global_load_async_to_lds_b128)
#define HAVE_ASYNC_LDS 1
typedef __attribute__((address_space(1))) v4i as1_v4i;
typedef __attribute__((address_space(3))) v4i as3_v4i;
#else
#define HAVE_ASYNC_LDS 0
#endif

__device__ __forceinline__ void copy16_g2lds(const bf16* gsrc, bf16* ldst) {
#if HAVE_ASYNC_LDS
  __builtin_amdgcn_global_load_async_to_lds_b128(
      (as1_v4i*)gsrc, (as3_v4i*)ldst, /*imm offset=*/0, /*cpol=*/0);
#else
  *(v4i*)ldst = *(const v4i*)gsrc;
#endif
}

__device__ __forceinline__ void wait_async_copies() {
#if HAVE_ASYNC_LDS
#if __has_builtin(__builtin_amdgcn_s_wait_asynccnt)
  __builtin_amdgcn_s_wait_asynccnt(0);
#else
  asm volatile("s_wait_asynccnt 0x0" ::: "memory");
#endif
#endif
}

// ---------------------------------------------------------------------------
// fp32 -> bf16 pre-conversion (8 elements / thread, 16B vector stores)
// ---------------------------------------------------------------------------
__global__ __launch_bounds__(256) void cvt_f32_to_bf16(const float* __restrict__ src,
                                                       bf16* __restrict__ dst, int n8) {
  const int i = blockIdx.x * blockDim.x + threadIdx.x;
  if (i >= n8) return;
  const v4f a = ((const v4f*)src)[2 * i];
  const v4f b = ((const v4f*)src)[2 * i + 1];
  v8bf o;
  #pragma unroll
  for (int j = 0; j < 4; ++j) { o[j] = (bf16)a[j]; o[4 + j] = (bf16)b[j]; }
  ((v8bf*)dst)[i] = o;
}

// ---------------------------------------------------------------------------
// GEMM: KF[m, n] = sum_d Xb[m, d] * Wb[n, d]   (bf16 in, fp32 out)
// BM=BN=128, BK=32. 8 waves; wave w owns rows w*16..w*16+15 and all 128 cols:
// 8 accumulators, 8 v_wmma per k-step sharing one A fragment.
// Double-buffered LDS fed by async global->LDS b128 copies.
// ---------------------------------------------------------------------------
__global__ __launch_bounds__(256) void gemm_kf(const bf16* __restrict__ Xb,
                                               const bf16* __restrict__ Wb,
                                               float* __restrict__ KF) {
  __shared__ __align__(16) bf16 As[2][BM][BK];   // 2 x 8 KB
  __shared__ __align__(16) bf16 Bs[2][BN][BK];   // 2 x 8 KB

  const int tid   = threadIdx.x;
  const int lane  = tid & 31;
  const int w     = tid >> 5;       // 0..7 : 16-row m-subtile
  const int khalf = lane >> 4;      // 0/1  : K-half group held by this lane
  const int lmod  = lane & 15;      // row (A) / col (B) within 16

  const int m0 = blockIdx.x * BM;
  const int n0 = blockIdx.y * BN;

  // staging: 128 rows per tile, 2 threads/row, 2 x 16B chunks per thread
  const int srow = tid >> 1;            // 0..127
  const int sch0 = (tid & 1) * 2;       // chunk base 0 or 2
  const bf16* Arow = Xb + (size_t)(m0 + srow) * D_IN;
  const bf16* Brow = Wb + (size_t)(n0 + srow) * D_IN;

  auto stage = [&](int buf, int k0) {
    #pragma unroll
    for (int c = 0; c < 2; ++c) {
      const int off = (sch0 + c) * 8;   // bf16 element offset (16B chunks)
      copy16_g2lds(Arow + k0 + off, &As[buf][srow][off]);
      copy16_g2lds(Brow + k0 + off, &Bs[buf][srow][off]);
    }
  };

  v8f acc[8] = {};

  stage(0, 0);
  wait_async_copies();
  __syncthreads();

  int cur = 0;
  for (int k0 = 0; k0 < D_IN; k0 += BK) {
    const int nxt = cur ^ 1;
    if (k0 + BK < D_IN) stage(nxt, k0 + BK);   // prefetch next tile (async)

    // A fragment (ISA 16-bit A 16x32 layout): lane holds row w*16+lmod;
    // halves[0:8] = K(8*khalf..), halves[8:16] = K(16+8*khalf..)
    v16bf afrag;
    {
      v8bf lo = *(const v8bf*)&As[cur][w * 16 + lmod][8 * khalf];
      v8bf hv = *(const v8bf*)&As[cur][w * 16 + lmod][16 + 8 * khalf];
      #pragma unroll
      for (int i = 0; i < 8; ++i) { afrag[i] = lo[i]; afrag[8 + i] = hv[i]; }
    }

    #pragma unroll
    for (int j = 0; j < 8; ++j) {
      v8bf lo = *(const v8bf*)&Bs[cur][j * 16 + lmod][8 * khalf];
      v8bf hv = *(const v8bf*)&Bs[cur][j * 16 + lmod][16 + 8 * khalf];
      v16bf bfrag;
      #pragma unroll
      for (int i = 0; i < 8; ++i) { bfrag[i] = lo[i]; bfrag[8 + i] = hv[i]; }
      acc[j] = __builtin_amdgcn_wmma_f32_16x16x32_bf16(
          false, afrag, false, bfrag, (short)0, acc[j], false, false);
    }

    wait_async_copies();   // own async copies into nxt done
    __syncthreads();       // publish nxt; everyone done reading cur
    cur = nxt;
  }

  // Epilogue: C/D layout — VGPR r: M = r + 8*(lane/16), N = lane%16
  #pragma unroll
  for (int j = 0; j < 8; ++j) {
    const int gn = n0 + j * 16 + lmod;
    #pragma unroll
    for (int r = 0; r < 8; ++r) {
      const int gm = m0 + w * 16 + 8 * khalf + r;
      KF[(size_t)gm * N_OUT + gn] = acc[j][r];
    }
  }
}

// ---------------------------------------------------------------------------
// Scan: one thread per (b, e) channel. c = sigmoid(-k), z = sigmoid(k),
// g = hx + 0.5 (hx>=0) else sigmoid(hx); h_t = c*h_{t-1} + z*g, h_0 = 0.5.
// ---------------------------------------------------------------------------
__global__ __launch_bounds__(256) void scan_kernel(const float* __restrict__ KF,
                                                   float* __restrict__ OUT) {
  const int idx = blockIdx.x * blockDim.x + threadIdx.x;   // 0..8191
  const int b = idx >> 10;
  const int e = idx & 1023;

  float h = 0.5f;
  size_t base  = (size_t)b * TSEQ * N_OUT + e;
  size_t obase = (size_t)b * TSEQ * D_IN + e;

  for (int t = 0; t < TSEQ; ++t) {
    const float k  = KF[base];
    const float hx = KF[base + D_IN];
    const float c  = 1.0f / (1.0f + __expf(k));     // exp(-softplus(k))
    const float z  = 1.0f - c;                      // sigmoid(k)
    const float g  = (hx >= 0.0f) ? (hx + 0.5f)
                                  : (1.0f / (1.0f + __expf(-hx)));
    h = fmaf(c, h, z * g);
    OUT[obase] = h;
    base  += N_OUT;
    obase += D_IN;
  }
}

extern "C" void kernel_launch(void* const* d_in, const int* in_sizes, int n_in,
                              void* d_out, int out_size, void* d_ws, size_t ws_size,
                              hipStream_t stream) {
  (void)in_sizes; (void)n_in; (void)out_size; (void)ws_size;
  const float* x  = (const float*)d_in[0];   // (8, 4096, 1024) fp32
  const float* Wf = (const float*)d_in[1];   // (2048, 1024) fp32
  float* out = (float*)d_out;                // (8, 4096, 1024) fp32

  // workspace layout: kf fp32 (256 MB) | Xb bf16 (64 MB) | Wb bf16 (4 MB)
  float* kf = (float*)d_ws;
  bf16* Xb = (bf16*)((char*)d_ws + (size_t)BT * N_OUT * sizeof(float));
  bf16* Wb = (bf16*)((char*)Xb + (size_t)BT * D_IN * sizeof(bf16));

  const int x8 = BT * D_IN / 8;      // 4,194,304 vec8 groups
  const int w8 = N_OUT * D_IN / 8;   //   262,144
  cvt_f32_to_bf16<<<x8 / 256, 256, 0, stream>>>(x,  Xb, x8);
  cvt_f32_to_bf16<<<w8 / 256, 256, 0, stream>>>(Wf, Wb, w8);

  dim3 g1(BT / BM, N_OUT / BN);      // 256 x 16 = 4096 workgroups
  gemm_kf<<<g1, 256, 0, stream>>>(Xb, Wb, kf);

  scan_kernel<<<(BATCH * D_IN) / 256, 256, 0, stream>>>(kf, out);
}
